// Sequence_73169062854680
// MI455X (gfx1250) — compile-verified, matrix operand order checked by reading, not run on hardware
//
#include <hip/hip_runtime.h>

typedef __attribute__((ext_vector_type(16))) __bf16 v16bf;
typedef __attribute__((ext_vector_type(8)))  float  v8f;
typedef unsigned short u16;

#define BB   512
#define TT   256
#define NI   64
#define NH   512
#define HSTR 520   // padded LDS row stride (elems): 260 DW -> bank-conflict-free
#define XSTR 72    // padded x row stride (elems): 36 DW -> bank-conflict-free
#define NTHR 512   // 16 waves; 4 waves/SIMD -> big VGPR budget, no spills

union V16U { uint4 q[2]; v16bf v; };

__device__ __forceinline__ u16 f2bf(float f) {
  __bf16 h = (__bf16)f;
  return __builtin_bit_cast(u16, h);
}
__device__ __forceinline__ float sigf(float x)     { return 1.0f / (1.0f + __expf(-x)); }
__device__ __forceinline__ float tanhfast(float x) { float e = __expf(-2.0f * x); return (1.0f - e) / (1.0f + e); }

__device__ __forceinline__ v8f zero8() {
  v8f v;
#pragma unroll
  for (int i = 0; i < 8; ++i) v[i] = 0.f;
  return v;
}

// One K-panel of the gate GEMM for one 16-column h-tile (4 gate blocks i/f/g/o).
// A: 16xK row-major bf16 (LDS).  B: pre-packed WMMA-B tiles (global, L2-resident).
template<int NKT>
__device__ __forceinline__ void gemm_part(v8f acc[4], const u16* __restrict__ wp, int wt,
                                          const u16* a_base, int astride, int lane) {
  const int row = lane & 15, hi = lane >> 4;
  const u16* ap = a_base + row * astride + hi * 8;
  const u16* bp[4];
#pragma unroll
  for (int g = 0; g < 4; ++g)
    bp[g] = wp + ((size_t)((g * 32 + wt) * NKT * 32 + lane) << 4);
#pragma unroll 4
  for (int kt = 0; kt < NKT; ++kt) {
    V16U a;
    a.q[0] = *(const uint4*)(ap + kt * 32);
    a.q[1] = *(const uint4*)(ap + kt * 32 + 16);
#pragma unroll
    for (int g = 0; g < 4; ++g) {
      V16U b;
      b.q[0] = *(const uint4*)(bp[g] + kt * 512);      // kt step = 1024B immediate
      b.q[1] = *(const uint4*)(bp[g] + kt * 512 + 8);
      acc[g] = __builtin_amdgcn_wmma_f32_16x16x32_bf16(false, a.v, false, b.v,
                                                       (short)0, acc[g], false, false);
    }
  }
}

// i/f/g/o (+bias) -> c,h update; h written to LDS (row-major bf16), f32 h
// returned for the head. Bias is folded here (scalar adds) so accumulators can
// start from the WMMA inline-0 C operand -> no splat registers, no spills.
__device__ __forceinline__ void cell_update(v8f acc[4], const float bia[4], v8f& c,
                                            u16* hdst, int m0, int n, float hv[8]) {
#pragma unroll
  for (int r = 0; r < 8; ++r) {
    float iv = sigf(acc[0][r] + bia[0]);
    float fv = sigf(acc[1][r] + bia[1]);
    float gv = tanhfast(acc[2][r] + bia[2]);
    float ov = sigf(acc[3][r] + bia[3]);
    float cn = fv * c[r] + iv * gv;
    c[r] = cn;
    float hn = ov * tanhfast(cn);
    hdst[(m0 + r) * HSTR + n] = f2bf(hn);
    hv[r] = hn;
  }
}

// Pack W (2048 x K, row-major f32) into bf16 WMMA-B tiles:
// dst[((ntile*K32 + ktile)*32 + lane)*16 + e]
__global__ void pack_w(const float* __restrict__ src, u16* __restrict__ dst, int K) {
  int p = blockIdx.x * blockDim.x + threadIdx.x;
  int e     = p & 15;
  int lane  = (p >> 4) & 31;
  int t32   = p >> 9;
  int K32   = K >> 5;
  int ktile = t32 % K32;
  int ntile = t32 / K32;
  int laneN = lane & 15, hi = lane >> 4;
  int nrow  = ntile * 16 + laneN;
  int kl    = (e < 8 ? e : e + 8) + hi * 8;
  int k     = ktile * 32 + kl;
  dst[p] = f2bf(src[(size_t)nrow * K + k]);
}

__global__ __launch_bounds__(NTHR, 1)
void lstm_persistent(const float* __restrict__ x,
                     const u16* __restrict__ Wih1, const u16* __restrict__ Whh1,
                     const u16* __restrict__ Wih2, const u16* __restrict__ Whh2,
                     const float* __restrict__ bih1, const float* __restrict__ bhh1,
                     const float* __restrict__ bih2, const float* __restrict__ bhh2,
                     const float* __restrict__ Wlin, const float* __restrict__ blin,
                     float* __restrict__ out) {
  __shared__ __align__(16) u16 h1buf[2][16 * HSTR];
  __shared__ __align__(16) u16 h2buf[2][16 * HSTR];
  __shared__ __align__(16) u16 xbuf[16 * XSTR];
  __shared__ float red0[16], red1[16];

  const int tid  = threadIdx.x;
  const int wid  = tid >> 5;         // 0..15: owns 32 h-columns (two 16-col tiles)
  const int lane = tid & 31;
  const int col  = lane & 15;
  const int m0   = (lane >> 4) * 8;  // D-layout row base for this lane
  const int b0   = blockIdx.x * 16;  // batch tile base

  for (int i = tid; i < 16 * HSTR; i += NTHR) {
    h1buf[0][i] = 0; h1buf[1][i] = 0;
    h2buf[0][i] = 0; h2buf[1][i] = 0;
  }
  if (tid < 16) { red0[tid] = 0.f; red1[tid] = 0.f; }

  float bia1[2][4], bia2[2][4], w0l[2], w1l[2];
#pragma unroll
  for (int ht = 0; ht < 2; ++ht) {
    const int nn = (wid * 2 + ht) * 16 + col;
#pragma unroll
    for (int g = 0; g < 4; ++g) {
      bia1[ht][g] = bih1[g * NH + nn] + bhh1[g * NH + nn];
      bia2[ht][g] = bih2[g * NH + nn] + bhh2[g * NH + nn];
    }
    w0l[ht] = Wlin[nn];
    w1l[ht] = Wlin[NH + nn];
  }
  const float bo0 = blin[0], bo1 = blin[1];

  v8f c1[2], c2[2];
  c1[0] = c1[1] = c2[0] = c2[1] = zero8();

  __syncthreads();

  int cur = 0;
  for (int t = 0; t < TT; ++t) {
    const int nxt = cur ^ 1;
    const bool last = (t == TT - 1);

    // Defeat LICM on the x-projection B-fragments (round 2: hoist -> per-step
    // scratch spill/reload) without destroying pointer provenance (round 3:
    // laundering the pointer degraded global_load -> flat_load). A laundered
    // zero SGPR offset keeps the address loop-variant while staying global.
    int lic0 = 0;
    asm volatile("" : "+s"(lic0));
    const u16* wih1_t = Wih1 + lic0;

    // stage x_t (16 x 64) as bf16 into LDS (2 elems/thread)
    {
#pragma unroll
      for (int s = 0; s < 2; ++s) {
        int idx = tid + s * NTHR;
        int m = idx >> 6, j = idx & 63;
        xbuf[m * XSTR + j] = f2bf(x[(size_t)(b0 + m) * (TT * NI) + t * NI + j]);
      }
    }
    __syncthreads();

    // ---------------- layer 1 : gates = x W_ih1^T + h1 W_hh1^T + b ----------------
#pragma unroll
    for (int ht = 0; ht < 2; ++ht) {
      const int wt = wid * 2 + ht;
      const int nn = wt * 16 + col;
      v8f acc[4];
#pragma unroll
      for (int g = 0; g < 4; ++g) acc[g] = zero8();   // -> WMMA inline-0 C
      gemm_part<2>(acc, wih1_t, wt, xbuf, XSTR, lane);
      gemm_part<16>(acc, Whh1, wt, h1buf[cur], HSTR, lane);
      float hv[8];
      cell_update(acc, bia1[ht], c1[ht], h1buf[nxt], m0, nn, hv);
    }
    __syncthreads();

    // ---------------- layer 2 : gates = h1' W_ih2^T + h2 W_hh2^T + b ----------------
    float p0[8], p1[8];
#pragma unroll
    for (int r = 0; r < 8; ++r) { p0[r] = 0.f; p1[r] = 0.f; }
#pragma unroll
    for (int ht = 0; ht < 2; ++ht) {
      const int wt = wid * 2 + ht;
      const int nn = wt * 16 + col;
      v8f acc[4];
#pragma unroll
      for (int g = 0; g < 4; ++g) acc[g] = zero8();   // -> WMMA inline-0 C
      gemm_part<16>(acc, Wih2, wt, h1buf[nxt], HSTR, lane);
      gemm_part<16>(acc, Whh2, wt, h2buf[cur], HSTR, lane);
      float hv[8];
      cell_update(acc, bia2[ht], c2[ht], h2buf[nxt], m0, nn, hv);
#pragma unroll
      for (int r = 0; r < 8; ++r) {
        p0[r] += hv[r] * w0l[ht];
        p1[r] += hv[r] * w1l[ht];
      }
    }
    // butterfly over the 16 lanes of each half (rows m0..m0+7 stay in-half)
#pragma unroll
    for (int off = 1; off < 16; off <<= 1)
#pragma unroll
      for (int r = 0; r < 8; ++r) p0[r] += __shfl_xor(p0[r], off);
    if (last) {
#pragma unroll
      for (int off = 1; off < 16; off <<= 1)
#pragma unroll
        for (int r = 0; r < 8; ++r) p1[r] += __shfl_xor(p1[r], off);
    }
    if (col == 0) {
#pragma unroll
      for (int r = 0; r < 8; ++r) atomicAdd(&red0[m0 + r], p0[r]);
      if (last)
#pragma unroll
        for (int r = 0; r < 8; ++r) atomicAdd(&red1[m0 + r], p1[r]);
    }
    __syncthreads();

    if (tid < 16) {
      out[(size_t)(b0 + tid) * (TT + 1) + t] = red0[tid] + bo0;
      red0[tid] = 0.f;
      if (last) out[(size_t)(b0 + tid) * (TT + 1) + TT] = red1[tid] + bo1;
    }
    cur = nxt;
  }
}

extern "C" void kernel_launch(void* const* d_in, const int* in_sizes, int n_in,
                              void* d_out, int out_size, void* d_ws, size_t ws_size,
                              hipStream_t stream) {
  const float* x    = (const float*)d_in[0];
  const float* Wih1 = (const float*)d_in[1];
  const float* Whh1 = (const float*)d_in[2];
  const float* bih1 = (const float*)d_in[3];
  const float* bhh1 = (const float*)d_in[4];
  const float* Wih2 = (const float*)d_in[5];
  const float* Whh2 = (const float*)d_in[6];
  const float* bih2 = (const float*)d_in[7];
  const float* bhh2 = (const float*)d_in[8];
  const float* Wlin = (const float*)d_in[9];
  const float* blin = (const float*)d_in[10];
  float* out = (float*)d_out;

  u16* ws    = (u16*)d_ws;
  u16* p_ih1 = ws;                        // 2048*64  bf16
  u16* p_hh1 = p_ih1 + 2048 * 64;         // 2048*512 bf16
  u16* p_ih2 = p_hh1 + 2048 * 512;
  u16* p_hh2 = p_ih2 + 2048 * 512;

  pack_w<<<(2048 * 64) / 256,  256, 0, stream>>>(Wih1, p_ih1, 64);
  pack_w<<<(2048 * 512) / 256, 256, 0, stream>>>(Whh1, p_hh1, 512);
  pack_w<<<(2048 * 512) / 256, 256, 0, stream>>>(Wih2, p_ih2, 512);
  pack_w<<<(2048 * 512) / 256, 256, 0, stream>>>(Whh2, p_hh2, 512);

  lstm_persistent<<<BB / 16, NTHR, 0, stream>>>(x, p_ih1, p_hh1, p_ih2, p_hh2,
                                                bih1, bhh1, bih2, bhh2,
                                                Wlin, blin, out);
}
